// Critic_32435593019725
// MI455X (gfx1250) — matrix operand order repaired and checked by reading, not compiled, not run on hardware
//
#include <hip/hip_runtime.h>

typedef __attribute__((ext_vector_type(16))) __bf16 v16bf;
typedef __attribute__((ext_vector_type(8)))  float  v8f;

#define BM 128
#define BN 128
#define BK 32
#define LDSS 40   // BK + 8 bf16 pad -> 80B row stride (16B aligned, bank-clean)

struct FragBF {
    union { v16bf v; struct { uint4 lo, hi; } u; };
};

// generic pointer -> LDS byte offset (address space 3)
__device__ __forceinline__ unsigned lds_off(const void* p) {
    return (unsigned)(unsigned long long)(uintptr_t)
        (__attribute__((address_space(3))) const void*)p;
}

// ---------------------------------------------------------------------------
// Tiled WMMA GEMM: C[M, ldc](+coff) = relu?(A[M,K] * BT[N,K]^T + bias[N])
// A, BT bf16 row-major; K multiple of 64 (so tile count K/32 is even);
// M mult of 128; N mult of 128.
// Double-buffered LDS fed by GLOBAL_LOAD_ASYNC_TO_LDS_B128 (ASYNCcnt),
// so no VGPR staging (and no scratch spills); copies overlap the WMMAs.
// ---------------------------------------------------------------------------
__global__ __launch_bounds__(256, 1) void gemm_bf16_wmma(
    const __bf16* __restrict__ A,
    const __bf16* __restrict__ BT,
    const float*  __restrict__ bias,
    __bf16* __restrict__ C,
    int K, int ldc, int coff, int relu)
{
    __shared__ __align__(16) __bf16 sA[2][BM][LDSS];
    __shared__ __align__(16) __bf16 sB[2][BN][LDSS];

    const int tid  = threadIdx.x;
    const int lane = tid & 31;
    const int wave = tid >> 5;
    const int wm   = wave & 1;        // 2 waves along M
    const int wn   = wave >> 1;       // 4 waves along N
    const int m0   = wm * 64;
    const int n0   = wn * 32;
    const int bm   = blockIdx.y * BM;
    const int bn   = blockIdx.x * BN;
    const int half = lane >> 4;
    const int l15  = lane & 15;

    v8f acc[4][2] = {};

    // Async staging: tile is 128x32 bf16 = 512 x 16B chunks; each of the 256
    // threads copies chunks (rowt, colt) and (rowt+64, colt) per matrix.
    const int rowt = tid >> 2;            // 0..63
    const int colt = (tid & 3) << 3;      // 0,8,16,24 (bf16 elems)
    const __bf16* gA = A  + (size_t)(bm + rowt) * K + colt;
    const __bf16* gB = BT + (size_t)(bn + rowt) * K + colt;
    const unsigned rstep = 64u * LDSS * 2;          // +64 rows, in bytes
    const unsigned ldsA0 = lds_off(&sA[0][rowt][colt]);
    const unsigned ldsA1 = lds_off(&sA[1][rowt][colt]);
    const unsigned ldsB0 = lds_off(&sB[0][rowt][colt]);
    const unsigned ldsB1 = lds_off(&sB[1][rowt][colt]);

    auto issue_tile = [&](int k0, int p) {
        const __bf16* a0 = gA + k0;
        const __bf16* b0 = gB + k0;
        unsigned la = p ? ldsA1 : ldsA0;
        unsigned lb = p ? ldsB1 : ldsB0;
        asm volatile("global_load_async_to_lds_b128 %0, %1, off"
                     :: "v"(la), "v"((unsigned long long)a0) : "memory");
        asm volatile("global_load_async_to_lds_b128 %0, %1, off"
                     :: "v"(lb), "v"((unsigned long long)b0) : "memory");
        asm volatile("global_load_async_to_lds_b128 %0, %1, off"
                     :: "v"(la + rstep), "v"((unsigned long long)(a0 + (size_t)64 * K)) : "memory");
        asm volatile("global_load_async_to_lds_b128 %0, %1, off"
                     :: "v"(lb + rstep), "v"((unsigned long long)(b0 + (size_t)64 * K)) : "memory");
    };

    auto compute = [&](const __bf16 (*cA)[LDSS], const __bf16 (*cB)[LDSS]) {
        FragBF af[4], bfr[2];
        #pragma unroll
        for (int im = 0; im < 4; im++) {
            // A 16x32 bf16: lane=M, lo: k=half*8+0..7, hi: k=16+half*8+0..7
            const __bf16* pa = &cA[m0 + im * 16 + l15][half * 8];
            af[im].u.lo = *(const uint4*)pa;
            af[im].u.hi = *(const uint4*)(pa + 16);
        }
        #pragma unroll
        for (int in = 0; in < 2; in++) {
            // B 32x16 bf16: lane=N, lo: k=half*8+0..7, hi: k=16+half*8+0..7
            const __bf16* pb = &cB[n0 + in * 16 + l15][half * 8];
            bfr[in].u.lo = *(const uint4*)pb;
            bfr[in].u.hi = *(const uint4*)(pb + 16);
        }
        #pragma unroll
        for (int im = 0; im < 4; im++)
            #pragma unroll
            for (int in = 0; in < 2; in++)
                acc[im][in] = __builtin_amdgcn_wmma_f32_16x16x32_bf16(
                    false, af[im].v, false, bfr[in].v,
                    (short)0, acc[im][in], false, false);
    };

    const int T = K / BK;                 // even for all our K
    issue_tile(0, 0);

    for (int kt = 0; kt < T; kt += 2) {
        // ---- even tile (buffer 0) ----
        issue_tile((kt + 1) * BK, 1);                     // kt+1 < T always
        asm volatile("s_wait_asynccnt 4" ::: "memory");   // tile kt landed
        __syncthreads();
        compute(sA[0], sB[0]);
        __syncthreads();
        // ---- odd tile (buffer 1) ----
        if (kt + 2 < T) {
            issue_tile((kt + 2) * BK, 0);
            asm volatile("s_wait_asynccnt 4" ::: "memory");
        } else {
            asm volatile("s_wait_asynccnt 0" ::: "memory");
        }
        __syncthreads();
        compute(sA[1], sB[1]);
        __syncthreads();
    }

    // Epilogue: bias + optional ReLU, bf16 store.
    // C/D layout: VGPR r, lanes0-15 -> M=r, N=lane; lanes16-31 -> M=r+8, N=lane-16
    #pragma unroll
    for (int im = 0; im < 4; im++) {
        #pragma unroll
        for (int in = 0; in < 2; in++) {
            int ncol = bn + n0 + in * 16 + l15;
            float bv = bias[ncol];
            #pragma unroll
            for (int r = 0; r < 8; r++) {
                int row = bm + m0 + im * 16 + half * 8 + r;
                float v = acc[im][in][r] + bv;
                if (relu) v = v > 0.f ? v : 0.f;
                C[(size_t)row * ldc + coff + ncol] = (__bf16)v;
            }
        }
    }
}

// ---------------------------------------------------------------------------
// fp32 [K,N] -> bf16 [N,Kpad] transpose+convert (zero-pad K..Kpad)
// ---------------------------------------------------------------------------
__global__ __launch_bounds__(256) void transpose_convert(
    const float* __restrict__ W, __bf16* __restrict__ WT,
    int K, int N, int Kpad)
{
    __shared__ float tile[32][33];
    int tx = threadIdx.x & 31;
    int ty = threadIdx.x >> 5;            // 0..7
    int kt = blockIdx.y * 32;
    int nt = blockIdx.x * 32;
    #pragma unroll
    for (int j = 0; j < 4; j++) {
        int k = kt + ty + j * 8;
        tile[ty + j * 8][tx] = (k < K) ? W[(size_t)k * N + nt + tx] : 0.f;
    }
    __syncthreads();
    #pragma unroll
    for (int j = 0; j < 4; j++) {
        int n = nt + ty + j * 8;
        int k = kt + tx;
        WT[(size_t)n * Kpad + k] = (__bf16)tile[tx][ty + j * 8];
    }
}

// fp32 [rows,scols] -> bf16 [rows,dcols], zero-pad extra columns
__global__ __launch_bounds__(256) void convert_pad(
    const float* __restrict__ src, __bf16* __restrict__ dst,
    int rows, int scols, int dcols)
{
    int idx = blockIdx.x * 256 + threadIdx.x;
    if (idx >= rows * dcols) return;
    int r = idx / dcols, c = idx % dcols;
    dst[idx] = (c < scols) ? (__bf16)src[(size_t)r * scols + c] : (__bf16)0.f;
}

// Prefill X[8192][576]: cols 0..7 = action (bf16), cols 520..575 = 0
__global__ __launch_bounds__(256) void fill_x(
    const float* __restrict__ action, __bf16* __restrict__ X)
{
    int idx = blockIdx.x * 256 + threadIdx.x;   // 8192*64 slots
    int r = idx >> 6, j = idx & 63;
    if (j < 8) X[(size_t)r * 576 + j] = (__bf16)action[r * 8 + j];
    else       X[(size_t)r * 576 + 520 + (j - 8)] = (__bf16)0.f;
}

// Final GEMV: q[row] = dot(H[row,:2048] bf16, w fp32) + b; one wave per row
__global__ __launch_bounds__(256) void gemv_final(
    const __bf16* __restrict__ H, const float* __restrict__ w,
    const float* __restrict__ b, float* __restrict__ q)
{
    int wave = threadIdx.x >> 5;
    int lane = threadIdx.x & 31;
    int row  = blockIdx.x * 8 + wave;
    const __bf16* hp = H + (size_t)row * 2048;
    float sum = 0.f;
    #pragma unroll
    for (int i = 0; i < 8; i++) {
        int base = i * 256 + lane * 8;
        uint4 hv = *(const uint4*)(hp + base);
        const __bf16* he = (const __bf16*)&hv;
        float4 w0 = *(const float4*)(w + base);
        float4 w1 = *(const float4*)(w + base + 4);
        sum += (float)he[0] * w0.x + (float)he[1] * w0.y +
               (float)he[2] * w0.z + (float)he[3] * w0.w;
        sum += (float)he[4] * w1.x + (float)he[5] * w1.y +
               (float)he[6] * w1.z + (float)he[7] * w1.w;
    }
    #pragma unroll
    for (int off = 16; off > 0; off >>= 1)
        sum += __shfl_xor(sum, off, 32);
    if (lane == 0) q[row] = sum + b[0];
}

extern "C" void kernel_launch(void* const* d_in, const int* in_sizes, int n_in,
                              void* d_out, int out_size, void* d_ws, size_t ws_size,
                              hipStream_t stream) {
    (void)in_sizes; (void)n_in; (void)out_size; (void)ws_size;
    const float* action = (const float*)d_in[0];
    const float* obs    = (const float*)d_in[1];
    const float* W1  = (const float*)d_in[2];  const float* b1  = (const float*)d_in[3];
    const float* W2  = (const float*)d_in[4];  const float* b2  = (const float*)d_in[5];
    const float* W3  = (const float*)d_in[6];  const float* b3  = (const float*)d_in[7];
    const float* Wm1 = (const float*)d_in[8];  const float* bm1 = (const float*)d_in[9];
    const float* Wm2 = (const float*)d_in[10]; const float* bm2 = (const float*)d_in[11];
    const float* Wm3 = (const float*)d_in[12]; const float* bm3 = (const float*)d_in[13];
    float* q = (float*)d_out;

    char* ws = (char*)d_ws;
    size_t off = 0;
    auto alloc = [&](size_t n) -> char* {
        char* p = ws + off;
        off += (n + 255) & ~(size_t)255;
        return p;
    };
    __bf16* Abf  = (__bf16*)alloc((size_t)8192 * 1024 * 2);
    __bf16* W1T  = (__bf16*)alloc((size_t)2048 * 1024 * 2);
    __bf16* W2T  = (__bf16*)alloc((size_t)2048 * 2048 * 2);
    __bf16* W3T  = (__bf16*)alloc((size_t)512  * 2048 * 2);
    __bf16* Wm1T = (__bf16*)alloc((size_t)2048 * 576  * 2);
    __bf16* Wm2T = (__bf16*)alloc((size_t)2048 * 2048 * 2);
    __bf16* H1   = (__bf16*)alloc((size_t)8192 * 2048 * 2);
    __bf16* H2   = (__bf16*)alloc((size_t)8192 * 2048 * 2);
    __bf16* X    = (__bf16*)alloc((size_t)8192 * 576  * 2);

    // --- one-time per-call conversions ---
    convert_pad<<<(8192 * 1024) / 256, 256, 0, stream>>>(obs, Abf, 8192, 1008, 1024);
    transpose_convert<<<dim3(2048/32, 1024/32), 256, 0, stream>>>(W1,  W1T,  1008, 2048, 1024);
    transpose_convert<<<dim3(2048/32, 2048/32), 256, 0, stream>>>(W2,  W2T,  2048, 2048, 2048);
    transpose_convert<<<dim3(512/32,  2048/32), 256, 0, stream>>>(W3,  W3T,  2048, 512,  2048);
    transpose_convert<<<dim3(2048/32, 576/32),  256, 0, stream>>>(Wm1, Wm1T, 520,  2048, 576);
    transpose_convert<<<dim3(2048/32, 2048/32), 256, 0, stream>>>(Wm2, Wm2T, 2048, 2048, 2048);
    fill_x<<<(8192 * 64) / 256, 256, 0, stream>>>(action, X);

    // --- WMMA GEMM chain ---
    gemm_bf16_wmma<<<dim3(2048/128, 8192/128), 256, 0, stream>>>(Abf, W1T,  b1,  H1, 1024, 2048, 0, 1);
    gemm_bf16_wmma<<<dim3(2048/128, 8192/128), 256, 0, stream>>>(H1,  W2T,  b2,  H2, 2048, 2048, 0, 1);
    gemm_bf16_wmma<<<dim3(512/128,  8192/128), 256, 0, stream>>>(H2,  W3T,  b3,  X,  2048, 576,  8, 0);
    gemm_bf16_wmma<<<dim3(2048/128, 8192/128), 256, 0, stream>>>(X,   Wm1T, bm1, H1, 576,  2048, 0, 1);
    gemm_bf16_wmma<<<dim3(2048/128, 8192/128), 256, 0, stream>>>(H1,  Wm2T, bm2, H2, 2048, 2048, 0, 1);

    // --- final 2048 -> 1 projection ---
    gemv_final<<<8192 / 8, 256, 0, stream>>>(H2, Wm3, bm3, q);
}
